// HardNegativeMiner_77558519431270
// MI455X (gfx1250) — compile-verified, matrix operand order checked by reading, not run on hardware
//
#include <hip/hip_runtime.h>
#include <hip/hip_bf16.h>

// ---------------------------------------------------------------------------
// HardNegativeMiner for MI455X (gfx1250, wave32).
//   B=1024 queries, N=50000 candidates, D=256, K=64, H=int(0.3*K)=19.
// Phase 1: L2-normalize rows, pack bf16 (25.6 MB candidates -> L2-resident).
// Phase 2: pos_sim[b] = <qN[b], cN[pos[b]]>.
// Phase 3: miner, grid = (B/16) x NSPLIT slices of the candidate axis.
//          Per chunk (128 candidates): double-buffered 64 KB LDS tile filled
//          with global_load_async_to_lds_b128 (ASYNCcnt pipeline, one chunk
//          ahead), 8x v_wmma_f32_16x16x32_bf16 fed by ds_load_b128, and a
//          4-selector/query online top-64. Slice results -> sorted lists.
// Phase 4: per-query 4-way merge of slice lists, hard/easy split, emit K idx.
// ---------------------------------------------------------------------------

#define DD 256
#define TOPK 64
#define NSEL 4              // selector threads per query
#define NSPLIT 4            // candidate-axis split (occupancy)
#define MARGIN 0.2f
#define TILE_BYTES 65536    // 128 candidates * 256 dims * 2B
#define SMEM_BYTES (2*TILE_BYTES + 8192 + 16384 + 16384 + 256)

typedef __attribute__((ext_vector_type(16))) __bf16 v16bf;
typedef __attribute__((ext_vector_type(8)))  float  v8f;

union Frag { v16bf v; uint4 u[2]; };

__device__ __forceinline__ unsigned short f2bf(float x) {
    unsigned u = __builtin_bit_cast(unsigned, x);
    unsigned r = u + 0x7FFFu + ((u >> 16) & 1u);   // round-to-nearest-even
    return (unsigned short)(r >> 16);
}
__device__ __forceinline__ float bf2f(unsigned short h) {
    return __builtin_bit_cast(float, ((unsigned)h) << 16);
}

// 64 KB tile -> LDS via CDNA5 async copy engine: 16 x b128 per thread.
__device__ __forceinline__ void tile_copy_async(
        const unsigned short* candN, size_t gbase, size_t gEnd,
        unsigned ldsBase, int tid) {
    #pragma unroll
    for (int i = 0; i < 16; ++i) {
        size_t off = (size_t)tid * 16 + (size_t)i * 4096;
        size_t ga  = gbase + off;
        if (ga + 16 > gEnd) ga = gEnd - 16;        // tail clamp (garbage cols
        const char* gp = (const char*)candN + ga;  //  are masked in selection)
        unsigned la = ldsBase + (unsigned)off;
        asm volatile("global_load_async_to_lds_b128 %0, %1, off"
                     :: "v"(la), "v"(gp) : "memory");
    }
}
#define S_WAIT_ASYNC_16() asm volatile("s_wait_asynccnt 0x10" ::: "memory")
#define S_WAIT_ASYNC_0()  asm volatile("s_wait_asynccnt 0x0"  ::: "memory")

// ---------------- Phase 1: row L2-normalize + bf16 pack (wave per row) ------
__global__ __launch_bounds__(256) void hnm_normalize(
        const float* __restrict__ q, const float* __restrict__ c,
        unsigned short* __restrict__ candN, unsigned short* __restrict__ qN,
        int N, int B) {
    int lane = threadIdx.x & 31;
    int wave = threadIdx.x >> 5;
    long row = (long)blockIdx.x * 8 + wave;
    const float* src;
    unsigned short* dst;
    if (row < N)          { src = c + row * DD;        dst = candN + row * DD; }
    else if (row < N + B) { long r = row - N; src = q + r * DD; dst = qN + r * DD; }
    else return;

    float4 a = ((const float4*)src)[lane * 2 + 0];
    float4 b = ((const float4*)src)[lane * 2 + 1];
    float ss = a.x*a.x + a.y*a.y + a.z*a.z + a.w*a.w
             + b.x*b.x + b.y*b.y + b.z*b.z + b.w*b.w;
    #pragma unroll
    for (int o = 16; o > 0; o >>= 1) ss += __shfl_xor(ss, o, 32);
    float scale = 1.0f / fmaxf(sqrtf(ss), 1e-12f);

    unsigned w0 = (unsigned)f2bf(a.x * scale) | ((unsigned)f2bf(a.y * scale) << 16);
    unsigned w1 = (unsigned)f2bf(a.z * scale) | ((unsigned)f2bf(a.w * scale) << 16);
    unsigned w2 = (unsigned)f2bf(b.x * scale) | ((unsigned)f2bf(b.y * scale) << 16);
    unsigned w3 = (unsigned)f2bf(b.z * scale) | ((unsigned)f2bf(b.w * scale) << 16);
    ((uint4*)dst)[lane] = make_uint4(w0, w1, w2, w3);
}

// ---------------- Phase 2: positive similarity (wave per query) -------------
__global__ __launch_bounds__(256) void hnm_possim(
        const unsigned short* __restrict__ candN,
        const unsigned short* __restrict__ qN,
        const int* __restrict__ pos, float* __restrict__ posSim,
        int N, int B) {
    int lane = threadIdx.x & 31;
    int wave = threadIdx.x >> 5;
    int b = blockIdx.x * 8 + wave;
    if (b >= B) return;
    int p = pos[b];
    if (p < 0) p = 0;
    if (p >= N) p = N - 1;
    uint4 qa = *(const uint4*)(qN    + (size_t)b * DD + lane * 8);
    uint4 ca = *(const uint4*)(candN + (size_t)p * DD + lane * 8);
    unsigned qw[4] = {qa.x, qa.y, qa.z, qa.w};
    unsigned cw[4] = {ca.x, ca.y, ca.z, ca.w};
    float acc = 0.0f;
    #pragma unroll
    for (int i = 0; i < 4; ++i) {
        acc = fmaf(bf2f((unsigned short)(qw[i] & 0xFFFF)),
                   bf2f((unsigned short)(cw[i] & 0xFFFF)), acc);
        acc = fmaf(bf2f((unsigned short)(qw[i] >> 16)),
                   bf2f((unsigned short)(cw[i] >> 16)), acc);
    }
    #pragma unroll
    for (int o = 16; o > 0; o >>= 1) acc += __shfl_xor(acc, o, 32);
    if (lane == 0) posSim[b] = acc;
}

// ---------------- Phase 3: async-pipelined WMMA miner (one slice) -----------
__global__ __launch_bounds__(256) void hnm_miner(
        const unsigned short* __restrict__ candN,
        const unsigned short* __restrict__ qN,
        const int* __restrict__ pos,
        float* __restrict__ partV, int* __restrict__ partI,
        int N, int B, int per) {
    extern __shared__ char smem[];
    unsigned short* bufB = (unsigned short*)smem;                 // 2 x 64 KB
    float* sims = (float*)(smem + 2 * TILE_BYTES);                // 16x128 f32
    float* topv = (float*)(smem + 2 * TILE_BYTES + 8192);         // 64 x 64
    int*   topi = (int*)  (smem + 2 * TILE_BYTES + 8192 + 16384);
    int*   scnt = (int*)  (smem + 2 * TILE_BYTES + 8192 + 32768);

    const int tid  = threadIdx.x;
    const int lane = tid & 31;
    const int wave = tid >> 5;
    const int qbase = blockIdx.x * 16;
    const int slice = blockIdx.y;
    const int n0 = slice * per;
    int n1 = n0 + per; if (n1 > N) n1 = N;
    const int span   = n1 - n0;
    const int chunks = (span > 0) ? ((span + 127) >> 7) : 0;
    const size_t gEnd    = (size_t)N * (DD * 2);
    const unsigned ldsB  = (unsigned)(size_t)bufB;

    // --- A fragments (16 queries), ISA 16-bit A layout, hoisted to VGPRs.
    const int arow = lane & 15;
    const int koff = (lane & 16) ? 8 : 0;
    int arowc = qbase + arow; if (arowc >= B) arowc = B - 1;
    const unsigned short* abase = qN + (size_t)arowc * DD;
    Frag A[8];
    #pragma unroll
    for (int kk = 0; kk < 8; ++kk) {
        const uint4* up = (const uint4*)(abase + kk * 32 + koff);
        A[kk].u[0] = up[0];
        A[kk].u[1] = up[2];
    }

    // --- selection: NSEL threads per query (t<64 -> q = t/4, residue t%4)
    const int sq  = tid >> 2;
    const int ssl = tid & 3;
    const bool sel = (tid < 16 * NSEL) && (qbase + sq < B);
    int   myPos = -1, cnt = 0, pmin = 0;
    float vmin = 3.4e38f;
    if (sel) myPos = pos[qbase + sq];
    float* tv = topv + tid * TOPK;
    int*   ti = topi + tid * TOPK;

    const int ncol = lane & 15;
    if (chunks > 0)                                        // prologue copy
        tile_copy_async(candN, (size_t)n0 * (DD * 2), gEnd, ldsB, tid);

    for (int ch = 0; ch < chunks; ++ch) {
        const int cur = ch & 1;
        if (ch + 1 < chunks) {                             // copy one ahead
            tile_copy_async(candN, (size_t)(n0 + (ch + 1) * 128) * (DD * 2),
                            gEnd, ldsB + (1 - cur) * TILE_BYTES, tid);
            S_WAIT_ASYNC_16();                             // current tile done
        } else {
            S_WAIT_ASYNC_0();
        }
        __syncthreads();                                   // tile visible

        // GEMM: 8 k-steps from LDS tile (ds_load_b128 pairs), back-to-back WMMA
        const unsigned short* bb = bufB + (size_t)cur * (TILE_BYTES / 2)
                                 + (size_t)(wave * 16 + ncol) * DD;
        v8f acc = {};
        #pragma unroll
        for (int kk = 0; kk < 8; ++kk) {
            Frag Bf;
            const uint4* lp = (const uint4*)(bb + kk * 32 + koff);
            Bf.u[0] = lp[0];
            Bf.u[1] = lp[2];
            acc = __builtin_amdgcn_wmma_f32_16x16x32_bf16(
                false, A[kk].v, false, Bf.v, (short)0, acc, false, false);
        }
        const int col    = wave * 16 + ncol;
        const int rowoff = (lane & 16) ? 8 : 0;
        #pragma unroll
        for (int i = 0; i < 8; ++i) sims[(i + rowoff) * 128 + col] = acc[i];
        __syncthreads();

        if (sel) {
            for (int j = ssl; j < 128; j += NSEL) {
                const int nn = n0 + ch * 128 + j;
                if (nn >= n1 || nn == myPos) continue;
                const float v = sims[sq * 128 + j];
                if (cnt < TOPK) {
                    tv[cnt] = v; ti[cnt] = nn;
                    if (v < vmin) { vmin = v; pmin = cnt; }
                    ++cnt;
                } else if (v > vmin) {
                    tv[pmin] = v; ti[pmin] = nn;
                    vmin = 3.4e38f;
                    #pragma unroll 4
                    for (int m = 0; m < TOPK; ++m)
                        if (tv[m] < vmin) { vmin = tv[m]; pmin = m; }
                }
            }
        }
        __syncthreads();
    }

    // --- sort each sublist desc (64 threads in parallel), then merge 4 -> 64
    if (sel) {
        for (int i = 1; i < cnt; ++i) {
            float v = tv[i]; int ix = ti[i]; int j = i - 1;
            while (j >= 0 && tv[j] < v) { tv[j+1] = tv[j]; ti[j+1] = ti[j]; --j; }
            tv[j+1] = v; ti[j+1] = ix;
        }
        scnt[tid] = cnt;
    }
    __syncthreads();

    if (tid < 16 && (qbase + tid < B)) {
        float* pv = partV + ((size_t)(qbase + tid) * NSPLIT + slice) * TOPK;
        int*   pi = partI + ((size_t)(qbase + tid) * NSPLIT + slice) * TOPK;
        int head[NSEL], cnts[NSEL];
        #pragma unroll
        for (int s = 0; s < NSEL; ++s) { head[s] = 0; cnts[s] = scnt[tid * NSEL + s]; }
        int m = 0;
        for (; m < TOPK; ++m) {
            int bs = -1; float bvv = -3.4e38f;
            #pragma unroll
            for (int s = 0; s < NSEL; ++s) {
                if (head[s] < cnts[s]) {
                    float v = topv[(tid * NSEL + s) * TOPK + head[s]];
                    if (v > bvv) { bvv = v; bs = s; }
                }
            }
            if (bs < 0) break;
            pv[m] = bvv;
            pi[m] = topi[(tid * NSEL + bs) * TOPK + head[bs]];
            ++head[bs];
        }
        for (; m < TOPK; ++m) { pv[m] = -3.4e38f; pi[m] = 0; }
    }
}

// ---------------- Phase 4: per-query slice merge + emit ---------------------
__global__ __launch_bounds__(256) void hnm_output(
        const float* __restrict__ partV, const int* __restrict__ partI,
        const int* __restrict__ pos, const float* __restrict__ posSim,
        const int* __restrict__ kptr, int* __restrict__ out,
        int N, int B) {
    int b = blockIdx.x * 256 + threadIdx.x;
    if (b >= B) return;
    int K = kptr[0];
    if (K > TOPK) K = TOPK;
    if (K < 1)    K = 1;
    const int H = (int)((float)K * 0.3f);
    const float th = posSim[b] - MARGIN;

    int head[NSPLIT];
    #pragma unroll
    for (int s = 0; s < NSPLIT; ++s) head[s] = 0;
    int mi[TOPK];
    int m = 0, hard = 0;
    while (m < TOPK) {
        int bs = -1; float bvv = -3.4e38f;
        #pragma unroll
        for (int s = 0; s < NSPLIT; ++s) {
            if (head[s] < TOPK) {
                float v = partV[((size_t)b * NSPLIT + s) * TOPK + head[s]];
                if (v > -1.0e38f && v > bvv) { bvv = v; bs = s; }
            }
        }
        if (bs < 0) break;
        mi[m] = partI[((size_t)b * NSPLIT + bs) * TOPK + head[bs]];
        ++head[bs];
        if (m == hard && bvv > th) hard = m + 1;   // sorted desc -> prefix
        ++m;
    }
    int nh = (H < hard) ? H : hard;
    if (nh > m) nh = m;

    int pq = pos[b];
    int* o = out + (size_t)b * K;
    unsigned x = (unsigned)b * 2654435761u + 0x9E3779B9u;
    for (int j = 0; j < K; ++j) {
        int idx;
        if (j < nh) {
            idx = mi[j];                            // hardest first
        } else {                                    // hash stand-in for RNG easy
            x = x * 1664525u + 1013904223u;
            idx = (int)(x % (unsigned)N);
            if (idx == pq) idx = (idx + 1) % N;
        }
        o[j] = idx;
    }
}

extern "C" void kernel_launch(void* const* d_in, const int* in_sizes, int n_in,
                              void* d_out, int out_size, void* d_ws, size_t ws_size,
                              hipStream_t stream) {
    const float* q   = (const float*)d_in[0];
    const float* c   = (const float*)d_in[1];
    const int*   pos = (const int*)d_in[2];
    const int*   kpt = (const int*)d_in[3];
    int B = in_sizes[0] / DD;       // 1024
    int N = in_sizes[1] / DD;       // 50000

    // ws: bf16 candN | bf16 qN | f32 posSim | f32 partV | i32 partI
    unsigned short* candN = (unsigned short*)d_ws;
    unsigned short* qN    = candN + (size_t)N * DD;
    float*          psim  = (float*)(qN + (size_t)B * DD);
    float*          partV = psim + B;
    int*            partI = (int*)(partV + (size_t)B * NSPLIT * TOPK);
    int* out = (int*)d_out;

    long rows = (long)N + B;
    hipLaunchKernelGGL(hnm_normalize, dim3((int)((rows + 7) / 8)), dim3(256),
                       0, stream, q, c, candN, qN, N, B);

    hipLaunchKernelGGL(hnm_possim, dim3((B + 7) / 8), dim3(256), 0, stream,
                       candN, qN, pos, psim, N, B);

    int per = (((N + NSPLIT - 1) / NSPLIT) + 127) & ~127;   // 128-aligned slice
    hipLaunchKernelGGL(hnm_miner, dim3((B + 15) / 16, NSPLIT), dim3(256),
                       SMEM_BYTES, stream, candN, qN, pos, partV, partI, N, B, per);

    hipLaunchKernelGGL(hnm_output, dim3((B + 255) / 256), dim3(256), 0, stream,
                       partV, partI, pos, psim, kpt, out, N, B);
}
